// Contrastive_loss_resnet_patches_46583215293141
// MI455X (gfx1250) — compile-verified, matrix operand order checked by reading, not run on hardware
//
#include <hip/hip_runtime.h>
#include <hip/hip_bf16.h>
#include <hip/hip_fp16.h>

typedef __attribute__((ext_vector_type(16))) _Float16 v16h;
typedef __attribute__((ext_vector_type(8)))  float    v8f;

#define BB 2      // batch
#define CC 256    // channels (GEMM K dim)
#define SS 512    // spatial positions D*H*W
#define PP 128    // patches
#define KK 512    // samples per patch (Gram matrix size)
#define INV_BW 20.0f   // 1 / 0.05
#define EPSV   1e-5f

// ---------------------------------------------------------------------------
// Kernel 1: L2-normalize over channels, transpose to [b][s][c], convert to f16.
// One wave32 per (b,s): 8 channels per lane, shuffle-reduce sum of squares.
// Also zeroes the global loss accumulator (deterministic per launch).
// ---------------------------------------------------------------------------
__global__ void k_norm(const float* __restrict__ src, _Float16* __restrict__ dst,
                       float* __restrict__ acc, int zero_acc) {
    if (zero_acc && blockIdx.x == 0 && threadIdx.x == 0) acc[0] = 0.0f;

    const int wave = threadIdx.x >> 5;
    const int lane = threadIdx.x & 31;
    const int pair = blockIdx.x * 8 + wave;       // grid = B*S/8 blocks of 8 waves
    const int b = pair / SS;
    const int s = pair % SS;

    const float* base = src + (size_t)b * CC * SS + s;   // channel stride = SS
    float x[8];
    float ss = 0.0f;
#pragma unroll
    for (int i = 0; i < 8; ++i) {
        x[i] = base[(size_t)(lane + 32 * i) * SS];
        ss += x[i] * x[i];
    }
#pragma unroll
    for (int off = 16; off > 0; off >>= 1)
        ss += __shfl_xor(ss, off, 32);            // wave32 butterfly reduction

    const float scale = 1.0f / fmaxf(sqrtf(ss), 1e-12f);

    _Float16* d = dst + ((size_t)b * SS + s) * CC;
#pragma unroll
    for (int i = 0; i < 8; ++i)
        d[lane + 32 * i] = (_Float16)(x[i] * scale);
}

// ---------------------------------------------------------------------------
// Kernel 2: per (p,b) pair: G = A2^T A1 (512x512, inner dim C=256) via WMMA.
// Register-blocked: each wave owns a 32x64 macro-tile (2 k-tiles x 4 l-tiles,
// 8 f32 accumulators). Per K-step: 6 fragment loads -> 8 wmma. The ~150-VGPR
// working set needs __launch_bounds__(256,1) so the compiler allocates
// registers instead of spilling to scratch (round-2 asm showed spills).
// Fused epilogue: exp(G/bw), locally-reduced LDS row/col sums + diagonal.
// ---------------------------------------------------------------------------
__global__ void __launch_bounds__(256, 1)
k_gram(const _Float16* __restrict__ t2n,
       const _Float16* __restrict__ t1n,
       const int* __restrict__ idx,
       float* __restrict__ acc) {
    __shared__ float rowsum[KK];   // sum over l of e[k,l]
    __shared__ float colsum[KK];   // sum over k of e[k,l]
    __shared__ float diag[KK];     // e[i,i]
    __shared__ float lsum;

    const int p = blockIdx.x / BB;
    const int b = blockIdx.x % BB;
    const int* ip = idx + (size_t)p * KK;
    __builtin_prefetch(ip, 0, 3);                 // global_prefetch_b8: idx row

    for (int i = threadIdx.x; i < KK; i += blockDim.x) {
        rowsum[i] = 0.0f; colsum[i] = 0.0f; diag[i] = 0.0f;
    }
    if (threadIdx.x == 0) lsum = 0.0f;
    __syncthreads();

    const int lane = threadIdx.x & 31;
    const int wave = threadIdx.x >> 5;            // 0..7
    const int n    = lane & 15;                   // column / row index in tile
    const int half = lane >> 4;                   // K-half selector (wave32 layout)

    const _Float16* a_base = t2n + (size_t)b * SS * CC;
    const _Float16* b_base = t1n + (size_t)b * SS * CC;

    // macro grid: 16 k-macros (32 rows) x 8 l-macros (64 cols) = 128 over 8 waves
    for (int mt = wave; mt < 16 * 8; mt += 8) {
        const int k0 = (mt >> 3) * 32;
        const int l0 = (mt & 7) * 64;

        // gathered spatial indices + channel-contiguous fragment pointers
        const v16h* ap[2];
        const v16h* bp[4];
#pragma unroll
        for (int ki = 0; ki < 2; ++ki) {
            const int sA = ip[k0 + ki * 16 + n];
            ap[ki] = (const v16h*)(a_base + (size_t)sA * CC + half * 16);
        }
#pragma unroll
        for (int li = 0; li < 4; ++li) {
            const int sB = ip[l0 + li * 16 + n];
            bp[li] = (const v16h*)(b_base + (size_t)sB * CC + half * 16);
        }

        v8f cacc[2][4] = {};
#pragma unroll
        for (int cc = 0; cc < 8; ++cc) {          // K-loop: 8 x (K=32) = C=256
            const v16h a0 = ap[0][2 * cc];
            const v16h a1 = ap[1][2 * cc];
            const v16h b0 = bp[0][2 * cc];
            const v16h b1 = bp[1][2 * cc];
            const v16h b2 = bp[2][2 * cc];
            const v16h b3 = bp[3][2 * cc];
            // 8 independent WMMAs hide the WMMA->WMMA hazard (disjoint D regs)
            cacc[0][0] = __builtin_amdgcn_wmma_f32_16x16x32_f16(
                false, a0, false, b0, (short)0, cacc[0][0], false, false);
            cacc[0][1] = __builtin_amdgcn_wmma_f32_16x16x32_f16(
                false, a0, false, b1, (short)0, cacc[0][1], false, false);
            cacc[0][2] = __builtin_amdgcn_wmma_f32_16x16x32_f16(
                false, a0, false, b2, (short)0, cacc[0][2], false, false);
            cacc[0][3] = __builtin_amdgcn_wmma_f32_16x16x32_f16(
                false, a0, false, b3, (short)0, cacc[0][3], false, false);
            cacc[1][0] = __builtin_amdgcn_wmma_f32_16x16x32_f16(
                false, a1, false, b0, (short)0, cacc[1][0], false, false);
            cacc[1][1] = __builtin_amdgcn_wmma_f32_16x16x32_f16(
                false, a1, false, b1, (short)0, cacc[1][1], false, false);
            cacc[1][2] = __builtin_amdgcn_wmma_f32_16x16x32_f16(
                false, a1, false, b2, (short)0, cacc[1][2], false, false);
            cacc[1][3] = __builtin_amdgcn_wmma_f32_16x16x32_f16(
                false, a1, false, b3, (short)0, cacc[1][3], false, false);
        }

        // fused epilogue: e = exp(G * 20); reduce locally across the macro-tile
        // before LDS atomics. C/D layout: VGPR r -> (M = r + 8*half, N = lane&15)
        float csum[4] = {0.0f, 0.0f, 0.0f, 0.0f};
#pragma unroll
        for (int ki = 0; ki < 2; ++ki) {
#pragma unroll
            for (int r = 0; r < 8; ++r) {
                const int k = k0 + ki * 16 + r + 8 * half;
                float rsum = 0.0f;
#pragma unroll
                for (int li = 0; li < 4; ++li) {
                    const float e = __expf(cacc[ki][li][r] * INV_BW);
                    rsum += e;
                    csum[li] += e;
                    if (k == l0 + li * 16 + n) diag[k] = e;
                }
                atomicAdd(&rowsum[k], rsum);      // ds_add_f32, 16 per macro-tile
            }
        }
#pragma unroll
        for (int li = 0; li < 4; ++li)
            atomicAdd(&colsum[l0 + li * 16 + n], csum[li]);  // 4 per macro-tile
    }
    __syncthreads();

    // per-(p,b) InfoNCE partial: -log(0.5*(d/colsum + d/rowsum) + eps)
    float t_loss = 0.0f;
    for (int i = threadIdx.x; i < KK; i += blockDim.x) {
        const float d = diag[i];
        const float v = 0.5f * (d / colsum[i] + d / rowsum[i]);
        t_loss += -__logf(v + EPSV);
    }
    atomicAdd(&lsum, t_loss);
    __syncthreads();
    if (threadIdx.x == 0) atomicAdd(acc, lsum);   // one global atomic per WG
}

// ---------------------------------------------------------------------------
// Kernel 3: finalize mean
// ---------------------------------------------------------------------------
__global__ void k_final(const float* __restrict__ acc, float* __restrict__ out) {
    if (blockIdx.x == 0 && threadIdx.x == 0)
        out[0] = acc[0] / (float)((size_t)PP * BB * KK);
}

extern "C" void kernel_launch(void* const* d_in, const int* in_sizes, int n_in,
                              void* d_out, int out_size, void* d_ws, size_t ws_size,
                              hipStream_t stream) {
    const float* t2  = (const float*)d_in[0];
    const float* t1  = (const float*)d_in[1];
    const int*   idx = (const int*)d_in[2];
    float* out = (float*)d_out;

    char* ws = (char*)d_ws;
    const size_t tensor_bytes = (size_t)BB * SS * CC * sizeof(_Float16);  // 512 KB
    _Float16* t2n = (_Float16*)ws;
    _Float16* t1n = (_Float16*)(ws + tensor_bytes);
    float*    acc = (float*)(ws + 2 * tensor_bytes);

    // 1024 (b,s) voxels, 8 waves per block
    k_norm<<<BB * SS / 8, 256, 0, stream>>>(t2, t2n, acc, 1);
    k_norm<<<BB * SS / 8, 256, 0, stream>>>(t1, t1n, acc, 0);

    // one workgroup per (p,b) Gram matrix
    k_gram<<<PP * BB, 256, 0, stream>>>(t2n, t1n, idx, acc);

    k_final<<<1, 32, 0, stream>>>(acc, out);
}
